// Reservoir_51771535786000
// MI455X (gfx1250) — matrix operand order repaired and checked by reading, not compiled, not run on hardware
//
#include <hip/hip_runtime.h>
#include <stdint.h>

// Problem constants (match reference setup_inputs)
constexpr int T = 512, B = 64, I = 256, H = 2048;
constexpr float ALPHA = 0.5f;

typedef __attribute__((ext_vector_type(16))) __bf16 v16bf;
typedef __attribute__((ext_vector_type(8)))  float  v8f;

union Frag { v16bf f; uint4 q[2]; };

// gfx1250 async global->LDS copy (ASYNCcnt path), if the toolchain exposes it.
#if defined(__has_builtin)
#if __has_builtin(__builtin_amdgcn_global_load_async_to_lds_b128)
#define HAVE_ASYNC_COPY 1
#endif
#endif
#ifndef HAVE_ASYNC_COPY
#define HAVE_ASYNC_COPY 0
#endif

#if HAVE_ASYNC_COPY
typedef int i4 __attribute__((ext_vector_type(4)));
typedef __attribute__((address_space(1))) i4 as1_i4;   // global
typedef __attribute__((address_space(3))) i4 as3_i4;   // LDS
#endif

__device__ __forceinline__ uint16_t f32_to_bf16(float x) {
    union { float f; uint32_t u; } v; v.f = x;
    uint32_t r = v.u + 0x7FFFu + ((v.u >> 16) & 1u);  // round-to-nearest-even
    return (uint16_t)(r >> 16);
}

// ---------------- conversion / init kernels ----------------

__global__ __launch_bounds__(256) void cvt_f32_to_bf16(const float* __restrict__ src,
                                                       uint16_t* __restrict__ dst, int n) {
    int i = blockIdx.x * 256 + threadIdx.x;
    if (i < n) dst[i] = f32_to_bf16(src[i]);
}

__global__ __launch_bounds__(256) void init_state(const float* __restrict__ s0,
                                                  float* __restrict__ sf,
                                                  uint16_t* __restrict__ sb, int n) {
    int i = blockIdx.x * 256 + threadIdx.x;
    if (i < n) { float v = s0[i]; sf[i] = v; sb[i] = f32_to_bf16(v); }
}

// ---------------- WMMA fragment loads (row-major source, stride ld) ----------------
// A 16x32 bf16: lane l -> row l&15; VGPR0-3 = K (l>>4)*8..+7, VGPR4-7 = +16.
__device__ __forceinline__ Frag load_a(const uint16_t* __restrict__ base, int lda,
                                       int m0, int k0, int lane) {
    int m = m0 + (lane & 15), hi = lane >> 4;
    const uint16_t* p = base + (size_t)m * lda + k0 + hi * 8;
    Frag fr;
    fr.q[0] = *(const uint4*)(p);
    fr.q[1] = *(const uint4*)(p + 16);
    return fr;
}
// B 32x16 bf16 from B^T row-major (W[n,k]): lane l -> col l&15; VGPR0-3 = K (l>>4)*8..+7, VGPR4-7 = +16.
__device__ __forceinline__ Frag load_b(const uint16_t* __restrict__ base, int ldb,
                                       int n0, int k0, int lane) {
    int n = n0 + (lane & 15), hi = lane >> 4;
    const uint16_t* p = base + (size_t)n * ldb + k0 + hi * 8;
    Frag fr;
    fr.q[0] = *(const uint4*)(p);
    fr.q[1] = *(const uint4*)(p + 16);
    return fr;
}

// ---------------- Phase 1: u = x @ W_in^T  (written into d_out) ----------------
// HBM-store-bound (256 MB of u); register-direct WMMA is sufficient here.
__global__ __launch_bounds__(256) void gemm_u(const uint16_t* __restrict__ Xb,   // [T*B, I] bf16
                                              const uint16_t* __restrict__ Winb, // [H, I]   bf16
                                              float* __restrict__ U) {           // [T*B, H] f32
    const int lane = threadIdx.x & 31;
    const int wave = threadIdx.x >> 5;
    const int m0 = blockIdx.x * 16;
    const int n0 = (blockIdx.y * 8 + wave) * 64;

    v8f acc0 = {}, acc1 = {}, acc2 = {}, acc3 = {};
    for (int k0 = 0; k0 < I; k0 += 32) {
        Frag a  = load_a(Xb, I, m0, k0, lane);
        Frag b0 = load_b(Winb, I, n0 +  0, k0, lane);
        Frag b1 = load_b(Winb, I, n0 + 16, k0, lane);
        Frag b2 = load_b(Winb, I, n0 + 32, k0, lane);
        Frag b3 = load_b(Winb, I, n0 + 48, k0, lane);
        acc0 = __builtin_amdgcn_wmma_f32_16x16x32_bf16(false, a.f, false, b0.f, (short)0, acc0, false, false);
        acc1 = __builtin_amdgcn_wmma_f32_16x16x32_bf16(false, a.f, false, b1.f, (short)0, acc1, false, false);
        acc2 = __builtin_amdgcn_wmma_f32_16x16x32_bf16(false, a.f, false, b2.f, (short)0, acc2, false, false);
        acc3 = __builtin_amdgcn_wmma_f32_16x16x32_bf16(false, a.f, false, b3.f, (short)0, acc3, false, false);
    }
    const int col = lane & 15, hi = lane >> 4;
    v8f* accs[4] = { &acc0, &acc1, &acc2, &acc3 };
#pragma unroll
    for (int j = 0; j < 4; ++j) {
#pragma unroll
        for (int r = 0; r < 8; ++r) {
            int m = m0 + r + 8 * hi;
            int n = n0 + 16 * j + col;
            U[(size_t)m * H + n] = (*accs[j])[r];
        }
    }
}

// ---------------- Phase 2: one recurrent step (LDS-staged, software-pipelined) ----
// Block = 256 threads (8 waves), owns N-strip of 128 and ALL of M=64, so W_hat is
// streamed from L2 exactly once per step. Double-buffered LDS stages of KC=64.
constexpr int KC = 64;                       // K per stage (2 WMMA k-chunks)
constexpr int A_ELEMS = 64 * KC;             // 8 KB
constexpr int B_ELEMS = 128 * KC;            // 16 KB
constexpr int STAGE_ELEMS = A_ELEMS + B_ELEMS;
constexpr int NSTAGE = H / KC;               // 32

struct FillRegs { uint4 a[2]; uint4 b[4]; };

__device__ __forceinline__ void async_copy16(const uint16_t* g, uint16_t* l) {
#if HAVE_ASYNC_COPY
    __builtin_amdgcn_global_load_async_to_lds_b128((as1_i4*)g, (as3_i4*)l, 0, 0);
#else
    (void)g; (void)l;
#endif
}

// Issue the global side of a stage fill. tid in [0,256).
__device__ __forceinline__ void issue_fill(const uint16_t* __restrict__ sb_in,
                                           const uint16_t* __restrict__ Whb,
                                           int n0, int k0, int tid,
                                           uint16_t* __restrict__ lbuf, FillRegs& fr) {
    // A tile: 64 rows x KC cols = 512 16B-chunks -> 2 per thread
#pragma unroll
    for (int j = 0; j < 2; ++j) {
        int c = tid + 256 * j;
        int row = c >> 3, col8 = (c & 7) * 8;
        const uint16_t* g = sb_in + (size_t)row * H + k0 + col8;
        uint16_t* l = lbuf + row * KC + col8;
#if HAVE_ASYNC_COPY
        async_copy16(g, l);
#else
        fr.a[j] = *(const uint4*)g; (void)l;
#endif
    }
    // B tile: 128 rows x KC cols = 1024 16B-chunks -> 4 per thread
#pragma unroll
    for (int j = 0; j < 4; ++j) {
        int c = tid + 256 * j;
        int row = c >> 3, col8 = (c & 7) * 8;
        const uint16_t* g = Whb + (size_t)(n0 + row) * H + k0 + col8;
        uint16_t* l = lbuf + A_ELEMS + row * KC + col8;
#if HAVE_ASYNC_COPY
        async_copy16(g, l);
#else
        fr.b[j] = *(const uint4*)g; (void)l;
#endif
    }
}

// Commit the LDS side of a stage fill (no-op on the async path).
__device__ __forceinline__ void commit_fill(int tid, uint16_t* __restrict__ lbuf,
                                            const FillRegs& fr) {
#if !HAVE_ASYNC_COPY
#pragma unroll
    for (int j = 0; j < 2; ++j) {
        int c = tid + 256 * j;
        int row = c >> 3, col8 = (c & 7) * 8;
        *(uint4*)(lbuf + row * KC + col8) = fr.a[j];
    }
#pragma unroll
    for (int j = 0; j < 4; ++j) {
        int c = tid + 256 * j;
        int row = c >> 3, col8 = (c & 7) * 8;
        *(uint4*)(lbuf + A_ELEMS + row * KC + col8) = fr.b[j];
    }
#else
    (void)tid; (void)lbuf; (void)fr;
#endif
}

__device__ __forceinline__ void fill_fence() {
#if HAVE_ASYNC_COPY
#if defined(__has_builtin) && __has_builtin(__builtin_amdgcn_s_wait_asynccnt)
    __builtin_amdgcn_s_wait_asynccnt(0);
#else
    asm volatile("s_wait_asynccnt 0" ::: "memory");
#endif
#endif
}

__global__ __launch_bounds__(256) void step_kernel(float* __restrict__ Ut,        // d_out + t*B*H
                                                   float* __restrict__ state_f,   // [B,H] f32 carry
                                                   const uint16_t* __restrict__ sb_in,  // [B,H] bf16
                                                   uint16_t* __restrict__ sb_out,       // [B,H] bf16
                                                   const uint16_t* __restrict__ Whb) {  // [H,H] bf16
    __shared__ uint16_t lds[2][STAGE_ELEMS];   // 48 KB of the 320 KB WGP pool
    const int tid  = threadIdx.x;
    const int lane = tid & 31;
    const int wave = tid >> 5;                 // owns n-tile `wave` of the strip
    const int n0   = blockIdx.x * 128;

    v8f acc0 = {}, acc1 = {}, acc2 = {}, acc3 = {};
    FillRegs fr;

    issue_fill(sb_in, Whb, n0, 0, tid, lds[0], fr);
    commit_fill(tid, lds[0], fr);
    fill_fence();
    __syncthreads();

    for (int s = 0; s < NSTAGE; ++s) {
        uint16_t* cur = lds[s & 1];
        if (s + 1 < NSTAGE)
            issue_fill(sb_in, Whb, n0, (s + 1) * KC, tid, lds[(s + 1) & 1], fr);

        const uint16_t* Al = cur;
        const uint16_t* Bl = cur + A_ELEMS;
#pragma unroll
        for (int kk = 0; kk < KC; kk += 32) {
            Frag b  = load_b(Bl, KC, 16 * wave, kk, lane);
            Frag a0 = load_a(Al, KC,  0, kk, lane);
            Frag a1 = load_a(Al, KC, 16, kk, lane);
            Frag a2 = load_a(Al, KC, 32, kk, lane);
            Frag a3 = load_a(Al, KC, 48, kk, lane);
            acc0 = __builtin_amdgcn_wmma_f32_16x16x32_bf16(false, a0.f, false, b.f, (short)0, acc0, false, false);
            acc1 = __builtin_amdgcn_wmma_f32_16x16x32_bf16(false, a1.f, false, b.f, (short)0, acc1, false, false);
            acc2 = __builtin_amdgcn_wmma_f32_16x16x32_bf16(false, a2.f, false, b.f, (short)0, acc2, false, false);
            acc3 = __builtin_amdgcn_wmma_f32_16x16x32_bf16(false, a3.f, false, b.f, (short)0, acc3, false, false);
        }

        if (s + 1 < NSTAGE)
            commit_fill(tid, lds[(s + 1) & 1], fr);
        fill_fence();
        __syncthreads();
    }

    // Epilogue: tanh + leaky blend; write f32 output, f32 carry, bf16 next-state.
    const int col = lane & 15, hi = lane >> 4;
    const int n = n0 + 16 * wave + col;
    v8f* accs[4] = { &acc0, &acc1, &acc2, &acc3 };
#pragma unroll
    for (int mt = 0; mt < 4; ++mt) {
#pragma unroll
        for (int r = 0; r < 8; ++r) {
            int m = 16 * mt + r + 8 * hi;
            size_t idx = (size_t)m * H + n;
            float pre = Ut[idx] + (*accs[mt])[r];
            float h   = tanhf(pre);
            float sv  = (1.0f - ALPHA) * state_f[idx] + ALPHA * h;
            Ut[idx]      = sv;
            state_f[idx] = sv;
            sb_out[idx]  = f32_to_bf16(sv);
        }
    }
}

// ---------------- host launcher ----------------
extern "C" void kernel_launch(void* const* d_in, const int* in_sizes, int n_in,
                              void* d_out, int out_size, void* d_ws, size_t ws_size,
                              hipStream_t stream) {
    const float* x   = (const float*)d_in[0];   // [T,B,I]
    const float* s0  = (const float*)d_in[1];   // [B,H]
    const float* Win = (const float*)d_in[2];   // [H,I]
    const float* Wh  = (const float*)d_in[3];   // [H,H]
    float* out = (float*)d_out;                 // [T,B,H]

    uint8_t* ws = (uint8_t*)d_ws;
    uint16_t* Whb  = (uint16_t*)ws;  ws += (size_t)H * H * 2;     // 8 MB (L2-resident)
    uint16_t* Winb = (uint16_t*)ws;  ws += (size_t)H * I * 2;     // 1 MB
    uint16_t* Xb   = (uint16_t*)ws;  ws += (size_t)T * B * I * 2; // 16 MB
    float*    Sf   = (float*)ws;     ws += (size_t)B * H * 4;     // 512 KB
    uint16_t* Sb0  = (uint16_t*)ws;  ws += (size_t)B * H * 2;     // 256 KB
    uint16_t* Sb1  = (uint16_t*)ws;                                // 256 KB

    cvt_f32_to_bf16<<<(H * H + 255) / 256, 256, 0, stream>>>(Wh, Whb, H * H);
    cvt_f32_to_bf16<<<(H * I + 255) / 256, 256, 0, stream>>>(Win, Winb, H * I);
    cvt_f32_to_bf16<<<(T * B * I + 255) / 256, 256, 0, stream>>>(x, Xb, T * B * I);
    init_state<<<(B * H + 255) / 256, 256, 0, stream>>>(s0, Sf, Sb0, B * H);

    // u = x @ W_in^T  -> d_out (consumed in place by the scan)
    gemm_u<<<dim3(T * B / 16, H / 512), 256, 0, stream>>>(Xb, Winb, out);

    // Sequential scan; kernel boundary = grid-wide sync between steps.
    for (int t = 0; t < T; ++t) {
        const uint16_t* sin  = (t & 1) ? Sb1 : Sb0;
        uint16_t*       sout = (t & 1) ? Sb0 : Sb1;
        step_kernel<<<dim3(H / 128), 256, 0, stream>>>(
            out + (size_t)t * B * H, Sf, sin, sout, Whb);
    }
}
